// Network_35072702939389
// MI455X (gfx1250) — matrix operand order repaired and checked by reading, not compiled, not run on hardware
//
#include <hip/hip_runtime.h>
#include <hip/hip_bf16.h>

#define N_NODES 50000
#define N_EDGES 800000
#define HDIM    128
#define NLAYERS 3
#define NGRAPH  512
#define NOUT    10
#define NPAD    50048   // N rounded up to multiple of 64 (GEMM row-tile)

typedef __attribute__((ext_vector_type(16))) __bf16       v16bf;
typedef __attribute__((ext_vector_type(8)))  float        v8f;
typedef __attribute__((ext_vector_type(4)))  unsigned int v4u;
typedef __attribute__((ext_vector_type(4)))  int          v4i;

union BFrag { v16bf v; v4u q[2]; };

// gfx1250 async global->LDS path (ASYNCcnt) with portable fallback.
// Probe-confirmed signature: (v4i AS1*, v4i AS3*, imm int offset, imm int cpol)
#if defined(__AMDGCN__) && __has_builtin(__builtin_amdgcn_global_load_async_to_lds_b128)
#define ASYNC_B2L 1
typedef __attribute__((address_space(1))) v4i* gas_v4i_ptr;
typedef __attribute__((address_space(3))) v4i* las_v4i_ptr;
#else
#define ASYNC_B2L 0
#endif

__device__ __forceinline__ void copy16_g2l(const unsigned short* g, unsigned short* l) {
#if ASYNC_B2L
  __builtin_amdgcn_global_load_async_to_lds_b128((gas_v4i_ptr)(g), (las_v4i_ptr)(l), 0, 0);
#else
  *reinterpret_cast<v4u*>(l) = *reinterpret_cast<const v4u*>(g);
#endif
}
__device__ __forceinline__ void wait_g2l() {
#if ASYNC_B2L
#if __has_builtin(__builtin_amdgcn_s_wait_asynccnt)
  __builtin_amdgcn_s_wait_asynccnt(0);
#else
  asm volatile("s_wait_asynccnt 0x0" ::: "memory");
#endif
#endif
}

// ---------- helpers ----------
__device__ __forceinline__ unsigned short f2bf(float f) {
  unsigned u = __float_as_uint(f);
  u += 0x7FFFu + ((u >> 16) & 1u);       // round-to-nearest-even
  return (unsigned short)(u >> 16);
}
// order-preserving float<->uint mapping for atomicMax-based segment max
__device__ __forceinline__ unsigned fmap(float f) {
  unsigned u = __float_as_uint(f);
  return (f >= 0.0f) ? (u | 0x80000000u) : ~u;
}
__device__ __forceinline__ float funmap(unsigned m) {
  unsigned u = (m & 0x80000000u) ? (m & 0x7FFFFFFFu) : ~m;
  return __uint_as_float(u);
}
struct __align__(8) US4 { unsigned short a, b, c, d; };
__device__ __forceinline__ void st_bf4(unsigned short* p, float x, float y, float z, float w) {
  US4 v{f2bf(x), f2bf(y), f2bf(z), f2bf(w)};
  *reinterpret_cast<US4*>(p) = v;
}

// ---------- small prep kernels ----------
__global__ void k_zero(float* __restrict__ p, long long n) {
  long long i = (long long)blockIdx.x * blockDim.x + threadIdx.x;
  if (i < n) p[i] = 0.0f;
}

__global__ void k_softmax_alphas(const float* __restrict__ lna, const float* __restrict__ lro,
                                 float* __restrict__ na_a, float* __restrict__ ro_a) {
  if (blockIdx.x || threadIdx.x) return;
  for (int i = 0; i < NLAYERS; ++i) {
    float m = -1e30f;
    for (int j = 0; j < 4; ++j) m = fmaxf(m, lna[i * 4 + j]);
    float e[4], s = 0.0f;
    for (int j = 0; j < 4; ++j) { e[j] = __expf(lna[i * 4 + j] - m); s += e[j]; }
    for (int j = 0; j < 4; ++j) na_a[i * 4 + j] = e[j] / s;
  }
  for (int i = 0; i < NLAYERS + 1; ++i) {
    float m = -1e30f;
    for (int j = 0; j < 3; ++j) m = fmaxf(m, lro[i * 3 + j]);
    float e[3], s = 0.0f;
    for (int j = 0; j < 3; ++j) { e[j] = __expf(lro[i * 3 + j] - m); s += e[j]; }
    for (int j = 0; j < 3; ++j) ro_a[i * 3 + j] = e[j] / s;
  }
}

// lin1 weight -> bf16, transposed to [n][k] (col-major for B-fragment loads)
__global__ void k_prep_lin1(const float* __restrict__ W, unsigned short* __restrict__ Bt) {
  int idx = blockIdx.x * blockDim.x + threadIdx.x;
  if (idx >= HDIM * HDIM) return;
  int n = idx & (HDIM - 1), k = idx >> 7;
  Bt[(size_t)n * HDIM + k] = f2bf(W[(size_t)k * HDIM + n]);
}

// fold the 5 per-layer matrices + softmaxed alphas into one K=512 weight,
// transposed: Wcat[i][n][q], q: 0..127 -> h-part, 128..255 -> A (gin),
// 256..383 -> A/deg (sage-nb), 384..511 -> gcn_agg
__global__ void k_prep_layers(const float* __restrict__ Wna, const float* __restrict__ na_a,
                              unsigned short* __restrict__ Wcat) {
  int idx = blockIdx.x * blockDim.x + threadIdx.x;
  if (idx >= NLAYERS * 4 * HDIM * HDIM) return;
  int i   = idx / (4 * HDIM * HDIM);
  int rem = idx - i * 4 * HDIM * HDIM;
  int q = rem >> 7;            // 0..511
  int n = rem & (HDIM - 1);
  const float* W = Wna + (size_t)i * 5 * HDIM * HDIM;
  const float* a = na_a + i * 4;
#define WM(j, k) W[((size_t)(j) * HDIM + (k)) * HDIM + n]
  float val;
  if (q < 128)       { int k = q;       val = a[1] * WM(1, k) + a[2] * WM(3, k) + a[3] * WM(4, k); }
  else if (q < 256)  { int k = q - 128; val = a[2] * WM(3, k); }
  else if (q < 384)  { int k = q - 256; val = a[1] * WM(2, k); }
  else               { int k = q - 384; val = a[0] * WM(0, k); }
#undef WM
  Wcat[((size_t)i * HDIM + n) * (4 * HDIM) + q] = f2bf(val);
}

__global__ void k_prep_out(const float* __restrict__ W, unsigned short* __restrict__ Bt) {
  int idx = blockIdx.x * blockDim.x + threadIdx.x;
  if (idx >= 4 * HDIM * HDIM) return;
  int n = idx & (HDIM - 1), k = idx >> 7;     // k in 0..511
  Bt[(size_t)n * (4 * HDIM) + k] = f2bf(W[(size_t)k * HDIM + n]);
}

__global__ void k_convert_x(const float* __restrict__ x, unsigned short* __restrict__ xbf) {
  int idx = blockIdx.x * blockDim.x + threadIdx.x;
  if (idx < N_NODES * HDIM) xbf[idx] = f2bf(x[idx]);
}

// ---------- graph-structure kernels ----------
__global__ void k_deg(const int* __restrict__ dst, float* __restrict__ deg) {
  int e = blockIdx.x * blockDim.x + threadIdx.x;
  if (e < N_EDGES) atomicAdd(&deg[dst[e]], 1.0f);
}
__global__ void k_cnt(const int* __restrict__ batch, float* __restrict__ cnt) {
  int n = blockIdx.x * blockDim.x + threadIdx.x;
  if (n < N_NODES) atomicAdd(&cnt[batch[n]], 1.0f);
}
__global__ void k_invdeg(const float* __restrict__ deg, float* __restrict__ invsq,
                         float* __restrict__ invdeg) {
  int n = blockIdx.x * blockDim.x + threadIdx.x;
  if (n >= N_NODES) return;
  float d = fmaxf(deg[n], 1.0f);
  invdeg[n] = 1.0f / d;
  invsq[n]  = rsqrtf(d);
}

// one wave per edge; one read of h[src] feeds both accumulators
__global__ void k_edge_aggregate(const int* __restrict__ src, const int* __restrict__ dst,
                                 const float* __restrict__ h, const float* __restrict__ invsq,
                                 float* __restrict__ Agg, float* __restrict__ Gc) {
  int e = blockIdx.x * 8 + (threadIdx.x >> 5);
  if (e >= N_EDGES) return;
  int lane = threadIdx.x & 31;
  int s = src[e], d = dst[e];
  float isq = invsq[s];
  float4 v = reinterpret_cast<const float4*>(h + (size_t)s * HDIM)[lane];
  float* ad = Agg + (size_t)d * HDIM + lane * 4;
  float* gd = Gc  + (size_t)d * HDIM + lane * 4;
  atomicAdd(ad + 0, v.x);       atomicAdd(ad + 1, v.y);
  atomicAdd(ad + 2, v.z);       atomicAdd(ad + 3, v.w);
  atomicAdd(gd + 0, v.x * isq); atomicAdd(gd + 1, v.y * isq);
  atomicAdd(gd + 2, v.z * isq); atomicAdd(gd + 3, v.w * isq);
}

// build bf16 Xcat[n] = [h | A | A/deg | Gc*deg^-1/2]  (row stride 512)
__global__ void k_build_xcat(const float* __restrict__ h, const float* __restrict__ Agg,
                             const float* __restrict__ Gc, const float* __restrict__ invdeg,
                             const float* __restrict__ invsq, unsigned short* __restrict__ X) {
  int idx = blockIdx.x * blockDim.x + threadIdx.x;
  int n = idx >> 5;
  if (n >= N_NODES) return;
  int lane = idx & 31, j = lane * 4;
  float4 hv = reinterpret_cast<const float4*>(h   + (size_t)n * HDIM)[lane];
  float4 av = reinterpret_cast<const float4*>(Agg + (size_t)n * HDIM)[lane];
  float4 gv = reinterpret_cast<const float4*>(Gc  + (size_t)n * HDIM)[lane];
  float idg = invdeg[n], isq = invsq[n];
  unsigned short* row = X + (size_t)n * (4 * HDIM);
  st_bf4(row +            j, hv.x, hv.y, hv.z, hv.w);
  st_bf4(row + 1 * HDIM + j, av.x, av.y, av.z, av.w);
  st_bf4(row + 2 * HDIM + j, av.x * idg, av.y * idg, av.z * idg, av.w * idg);
  st_bf4(row + 3 * HDIM + j, gv.x * isq, gv.y * isq, gv.z * isq, gv.w * isq);
}

// ---------- WMMA GEMM: C[M x NC] = act(A[M x K](bf16) @ Bt^T + bias) ----------
// A row-major bf16 (rows padded to mult of 64), Bt is B transposed: [NC][K] bf16.
// Block = 128 threads = 4 waves; each wave: 16 rows x NC cols.
// Per k-step the 32xNC B-slice is staged in LDS (async global->LDS when
// available) and consumed via conflict-free ds_load_b128 fragment reads.
#define BS_STRIDE 40   // halves per column slice (80B: 16B-aligned, bank-conflict-free)
template <int NC>
__global__ void __launch_bounds__(128)
k_wmma_gemm(const unsigned short* __restrict__ A, const unsigned short* __restrict__ Bt,
            const float* __restrict__ bias, float* __restrict__ C,
            int M, int K, int do_relu) {
  static_assert(NC == 128, "column-copy mapping assumes NC==128");
  __shared__ unsigned short Bs[NC * BS_STRIDE];   // 10 KB

  const int tid  = threadIdx.x;        // 0..127 == B column owned for staging
  const int lane = threadIdx.x & 31;
  const int wave = threadIdx.x >> 5;
  const int m0   = blockIdx.x * 64 + wave * 16;
  const int l15  = lane & 15;
  const int hi   = lane >> 4;
  const long long arow = m0 + l15;
  const int akoff = hi * 8;    // A: lanes 16-31 hold K+8..15 / K+24..31
  const int bkoff = hi * 8;    // B: lanes 16-31 hold K+8..15 / K+24..31

  v8f acc[NC / 16] = {};
  for (int k0 = 0; k0 < K; k0 += 32) {
    __syncthreads();                       // previous slice fully consumed
    {
      const unsigned short* gsrc = Bt + (long long)tid * K + k0;   // 64B per col
      unsigned short* ldst = &Bs[tid * BS_STRIDE];
#pragma unroll
      for (int i = 0; i < 4; ++i) copy16_g2l(gsrc + i * 8, ldst + i * 8);
    }
    wait_g2l();
    __syncthreads();                       // slice visible to all waves

    BFrag a;
    const unsigned short* ap = A + arow * K + k0 + akoff;
    a.q[0] = *reinterpret_cast<const v4u*>(ap);
    a.q[1] = *reinterpret_cast<const v4u*>(ap + 16);
    __builtin_prefetch(ap + 32, 0, 1);
#pragma unroll
    for (int t = 0; t < NC / 16; ++t) {
      BFrag b;
      const unsigned short* bp = &Bs[(t * 16 + l15) * BS_STRIDE + bkoff];
      b.q[0] = *reinterpret_cast<const v4u*>(bp);
      b.q[1] = *reinterpret_cast<const v4u*>(bp + 16);
      acc[t] = __builtin_amdgcn_wmma_f32_16x16x32_bf16(
          false, a.v, false, b.v, (short)0, acc[t], false, false);
    }
  }
#pragma unroll
  for (int t = 0; t < NC / 16; ++t) {
    int c = t * 16 + l15;
    float bv = bias ? bias[c] : 0.0f;
#pragma unroll
    for (int r = 0; r < 8; ++r) {
      int row = m0 + r + hi * 8;   // C layout: lanes 0-15 -> M=r, 16-31 -> M=r+8
      if (row < M) {
        float v = acc[t][r] + bv;
        if (do_relu) v = fmaxf(v, 0.0f);
        C[(long long)row * NC + c] = v;
      }
    }
  }
}

// ---------- readout ----------
__global__ void k_readout_acc(const float* __restrict__ h, const int* __restrict__ batch,
                              float* __restrict__ sums, unsigned* __restrict__ mx) {
  int idx = blockIdx.x * blockDim.x + threadIdx.x;
  if (idx >= N_NODES * HDIM) return;
  int n = idx >> 7, j = idx & (HDIM - 1);
  int g = batch[n];
  float f = h[idx];
  atomicAdd(&sums[(size_t)g * HDIM + j], f);
  atomicMax(&mx[(size_t)g * HDIM + j], fmap(f));
}

__global__ void k_readout_combine(const float* __restrict__ sums, const unsigned* __restrict__ mx,
                                  const float* __restrict__ cnt, const float* __restrict__ ro_a,
                                  int r, unsigned short* __restrict__ gbf) {
  int idx = blockIdx.x * blockDim.x + threadIdx.x;
  if (idx >= NGRAPH * HDIM) return;
  int gi = idx >> 7, j = idx & (HDIM - 1);
  const float* w = ro_a + r * 3;
  float c = cnt[gi];
  float s = sums[idx];
  float mean = s / fmaxf(c, 1.0f);
  float m = (c > 0.0f) ? funmap(mx[idx]) : 0.0f;
  gbf[(size_t)gi * (4 * HDIM) + r * HDIM + j] = f2bf(w[0] * mean + w[1] * m + w[2] * s);
}

// ---------- classifier ----------
__global__ void k_cls(const float* __restrict__ gfeat, const float* __restrict__ W,
                      const float* __restrict__ b, float* __restrict__ out) {
  int idx = blockIdx.x * blockDim.x + threadIdx.x;
  if (idx >= NGRAPH * NOUT) return;
  int gi = idx / NOUT, o = idx - gi * NOUT;
  float acc = b[o];
  const float* gr = gfeat + (size_t)gi * HDIM;
#pragma unroll 8
  for (int k = 0; k < HDIM; ++k) acc += gr[k] * W[(size_t)k * NOUT + o];
  out[idx] = acc;
}

// ---------- launch ----------
extern "C" void kernel_launch(void* const* d_in, const int* in_sizes, int n_in,
                              void* d_out, int out_size, void* d_ws, size_t ws_size,
                              hipStream_t stream) {
  (void)in_sizes; (void)n_in; (void)out_size; (void)ws_size;
  const float* x      = (const float*)d_in[0];
  const int*   ei     = (const int*)d_in[1];
  const int*   batch  = (const int*)d_in[2];
  const float* lin1_W = (const float*)d_in[3];
  const float* lin1_b = (const float*)d_in[4];
  const float* Wna    = (const float*)d_in[5];
  const float* lna    = (const float*)d_in[6];
  const float* lro    = (const float*)d_in[7];
  const float* lout_W = (const float*)d_in[8];
  const float* lout_b = (const float*)d_in[9];
  const float* cls_W  = (const float*)d_in[10];
  const float* cls_b  = (const float*)d_in[11];
  float* out = (float*)d_out;
  const int* src = ei;
  const int* dst = ei + N_EDGES;

  char* ws = (char*)d_ws;
  size_t off = 0;
  auto alloc = [&](size_t bytes) -> void* {
    void* p = ws + off;
    off = (off + bytes + 255) & ~(size_t)255;
    return p;
  };
  float*          h      = (float*)alloc((size_t)N_NODES * HDIM * 4);
  float*          Agg    = (float*)alloc((size_t)N_NODES * HDIM * 4);
  float*          Gc     = (float*)alloc((size_t)N_NODES * HDIM * 4);
  float*          deg    = (float*)alloc((size_t)N_NODES * 4);
  float*          invsq  = (float*)alloc((size_t)N_NODES * 4);
  float*          invdeg = (float*)alloc((size_t)N_NODES * 4);
  unsigned short* xbf    = (unsigned short*)alloc((size_t)NPAD * HDIM * 2);
  unsigned short* Xcat   = (unsigned short*)alloc((size_t)NPAD * 4 * HDIM * 2);
  unsigned short* Wl1t   = (unsigned short*)alloc((size_t)HDIM * HDIM * 2);
  unsigned short* Wcat   = (unsigned short*)alloc((size_t)NLAYERS * 4 * HDIM * HDIM * 2);
  unsigned short* Woutt  = (unsigned short*)alloc((size_t)4 * HDIM * HDIM * 2);
  float*          alph   = (float*)alloc(32 * 4);   // na_a[12] | ro_a[12]
  float*          sums   = (float*)alloc((size_t)NGRAPH * HDIM * 4);
  unsigned*       mx     = (unsigned*)alloc((size_t)NGRAPH * HDIM * 4);
  float*          cnt    = (float*)alloc((size_t)NGRAPH * 4);
  unsigned short* gbf    = (unsigned short*)alloc((size_t)NGRAPH * 4 * HDIM * 2);
  float*          gfeat  = (float*)alloc((size_t)NGRAPH * HDIM * 4);

  const int T = 256;
  float* na_a = alph;
  float* ro_a = alph + 12;

  // alphas + weight folding + input conversion
  k_softmax_alphas<<<1, 32, 0, stream>>>(lna, lro, na_a, ro_a);
  k_prep_lin1<<<(HDIM * HDIM + T - 1) / T, T, 0, stream>>>(lin1_W, Wl1t);
  k_prep_layers<<<(NLAYERS * 4 * HDIM * HDIM + T - 1) / T, T, 0, stream>>>(Wna, na_a, Wcat);
  k_prep_out<<<(4 * HDIM * HDIM + T - 1) / T, T, 0, stream>>>(lout_W, Woutt);
  k_convert_x<<<(N_NODES * HDIM + T - 1) / T, T, 0, stream>>>(x, xbf);

  // degrees & per-graph counts (re-zeroed every launch: graph-replay safe)
  k_zero<<<(N_NODES + T - 1) / T, T, 0, stream>>>(deg, N_NODES);
  k_zero<<<(NGRAPH + T - 1) / T, T, 0, stream>>>(cnt, NGRAPH);
  k_deg<<<(N_EDGES + T - 1) / T, T, 0, stream>>>(dst, deg);
  k_cnt<<<(N_NODES + T - 1) / T, T, 0, stream>>>(batch, cnt);
  k_invdeg<<<(N_NODES + T - 1) / T, T, 0, stream>>>(deg, invsq, invdeg);

  const int MB = (N_NODES + 63) / 64;

  // h = x @ lin1_W + b
  k_wmma_gemm<HDIM><<<MB, 128, 0, stream>>>(xbf, Wl1t, lin1_b, h, N_NODES, HDIM, 0);

  auto readout = [&](int r) {
    int GB = (NGRAPH * HDIM + T - 1) / T;
    k_zero<<<GB, T, 0, stream>>>(sums, (long long)NGRAPH * HDIM);
    k_zero<<<GB, T, 0, stream>>>((float*)mx, (long long)NGRAPH * HDIM);
    k_readout_acc<<<(N_NODES * HDIM + T - 1) / T, T, 0, stream>>>(h, batch, sums, mx);
    k_readout_combine<<<GB, T, 0, stream>>>(sums, mx, cnt, ro_a, r, gbf);
  };
  readout(0);

  for (int i = 0; i < NLAYERS; ++i) {
    long long NH = (long long)N_NODES * HDIM;
    int ZB = (int)((NH + T - 1) / T);
    k_zero<<<ZB, T, 0, stream>>>(Agg, NH);
    k_zero<<<ZB, T, 0, stream>>>(Gc, NH);
    k_edge_aggregate<<<(N_EDGES + 7) / 8, 256, 0, stream>>>(src, dst, h, invsq, Agg, Gc);
    k_build_xcat<<<(N_NODES * 32 + T - 1) / T, T, 0, stream>>>(h, Agg, Gc, invdeg, invsq, Xcat);
    // fused 4-in-1 GEMM (K=512) + ReLU, overwrites h
    k_wmma_gemm<HDIM><<<MB, 128, 0, stream>>>(
        Xcat, Wcat + (size_t)i * 4 * HDIM * HDIM, nullptr, h, N_NODES, 4 * HDIM, 1);
    readout(i + 1);
  }

  // g = relu(concat_readouts @ lin_out_W + b)  -> [512,128]
  k_wmma_gemm<HDIM><<<(NGRAPH + 63) / 64, 128, 0, stream>>>(
      gbf, Woutt, lout_b, gfeat, NGRAPH, 4 * HDIM, 1);
  // out = g @ cls_W + cls_b  -> [512,10]
  k_cls<<<(NGRAPH * NOUT + T - 1) / T, T, 0, stream>>>(gfeat, cls_W, cls_b, out);
}
